// MLPBlock_58205396795482
// MI455X (gfx1250) — compile-verified, matrix operand order checked by reading, not run on hardware
//
#include <hip/hip_runtime.h>
#include <hip/hip_bf16.h>
#include <cstdint>

// ---------------------------------------------------------------------------
// MoE MLP block for MI455X (gfx1250, wave32, WMMA).
// bf16 WMMA (16x16x32) with fp32 accumulation; fp32 weights converted to bf16
// during LDS staging (weights must stream from HBM anyway: ~800MB/call,
// ~34us @ 23.3TB/s; bf16 tensor compute dominates at ~100-200us).
// Round 4: fix async builtin pointer types (int* in AS1/AS3). Otherwise:
// all fragments loaded up-front per K-step (staggered dscnt waits, WMMA
// burst), async global->LDS staging for the bf16 activation tile in GEMM2
// (ASYNCcnt path), double-buffered LDS, weight-stream prefetch.
// ---------------------------------------------------------------------------

#define HIDDEN 2880
#define INTER  2880
#define NEXP   8
#define TOPK   4
#define NTOK   2048
#define ALPHA  1.702f
#define BETA   1.0f
#define LIMIT  7.0f

#define BM   128   // rows per workgroup tile (4 waves x 32 rows)
#define BN   96    // cols per workgroup tile (2 waves x 48 cols)
#define BK   32    // K step (one bf16 WMMA K)
#define LDSS 40    // LDS row stride in bf16 (80B: 16B-aligned, bank-spread)

typedef __bf16 bf16_t;
typedef __attribute__((ext_vector_type(2)))  __bf16 bhalf2;
typedef __attribute__((ext_vector_type(8)))  __bf16 bhalf8;
typedef __attribute__((ext_vector_type(16))) __bf16 bhalf16;
typedef __attribute__((ext_vector_type(8)))  float  float8;

typedef __attribute__((address_space(1))) int gint_t;   // global int
typedef __attribute__((address_space(3))) int lint_t;   // LDS int

#if __has_builtin(__builtin_amdgcn_global_load_async_to_lds_b32)
#define USE_ASYNC_LDS 1
#else
#define USE_ASYNC_LDS 0
#endif

__device__ __forceinline__ void wait_async0() {
#if USE_ASYNC_LDS
#if __has_builtin(__builtin_amdgcn_s_wait_asynccnt)
    __builtin_amdgcn_s_wait_asynccnt(0);
#else
    asm volatile("s_wait_asynccnt 0" ::: "memory");
#endif
#endif
}

// Build a v16bf fragment from two 16-byte LDS chunks.
// ISA 16-bit A/B layout (16x32): lane L holds row/col (L&15); half = L>>4;
// VGPRs 0..3 hold K = 8*half + 0..7, VGPRs 4..7 hold K = 16 + 8*half + 0..7.
__device__ __forceinline__ bhalf16 make_frag(const bf16_t* p) {
    bhalf8 lo = *(const bhalf8*)p;         // K = 8*half + 0..7
    bhalf8 hi = *(const bhalf8*)(p + 16);  // K = 16 + 8*half + 0..7
    return __builtin_shufflevector(lo, hi, 0, 1, 2, 3, 4, 5, 6, 7,
                                           8, 9, 10, 11, 12, 13, 14, 15);
}

__device__ __forceinline__ float8 wmma_bf16(bhalf16 a, bhalf16 b, float8 c) {
    return __builtin_amdgcn_wmma_f32_16x16x32_bf16(false, a, false, b,
                                                   (short)0, c, false, false);
}

__device__ __forceinline__ bhalf2 pack2(float a, float b) {
    bhalf2 h; h[0] = (bf16_t)a; h[1] = (bf16_t)b; return h;
}

// ---------------------------------------------------------------------------
// Router: one wave32 per token. logits = x @ gate_w + gate_b ; top-4 ;
// softmax over selected ; dense combine-weight row [8] written to cw.
// ---------------------------------------------------------------------------
__global__ __launch_bounds__(256) void moe_router_kernel(
    const float* __restrict__ x, const float* __restrict__ gw,
    const float* __restrict__ gb, float* __restrict__ cw)
{
    const int token = blockIdx.x * 8 + (threadIdx.x >> 5);
    const int lane  = threadIdx.x & 31;
    const float* xr = x + (size_t)token * HIDDEN;

    float p[NEXP];
#pragma unroll
    for (int e = 0; e < NEXP; ++e) p[e] = 0.0f;

    for (int h = lane; h < HIDDEN; h += 32) {
        const float xv = xr[h];
        const float* g = gw + (size_t)h * NEXP;
#pragma unroll
        for (int e = 0; e < NEXP; ++e) p[e] += xv * g[e];
    }
#pragma unroll
    for (int e = 0; e < NEXP; ++e) {
        float v = p[e];
#pragma unroll
        for (int off = 16; off > 0; off >>= 1) v += __shfl_down(v, off, 32);
        p[e] = v;
    }
    if (lane == 0) {
        float logit[NEXP];
#pragma unroll
        for (int e = 0; e < NEXP; ++e) logit[e] = p[e] + gb[e];
        unsigned used = 0;
        int   ci[TOPK];
        float cv[TOPK];
#pragma unroll
        for (int k = 0; k < TOPK; ++k) {
            int bi = 0; float bv = -1e30f;
#pragma unroll
            for (int e = 0; e < NEXP; ++e) {
                bool ok = !((used >> e) & 1u) && (logit[e] > bv);
                bv = ok ? logit[e] : bv;
                bi = ok ? e : bi;
            }
            used |= 1u << bi; ci[k] = bi; cv[k] = bv;
        }
        const float m = cv[0];
        float s = 0.0f, w[TOPK];
#pragma unroll
        for (int k = 0; k < TOPK; ++k) { w[k] = __expf(cv[k] - m); s += w[k]; }
        const float inv = 1.0f / s;
        float outw[NEXP];
#pragma unroll
        for (int e = 0; e < NEXP; ++e) outw[e] = 0.0f;
#pragma unroll
        for (int k = 0; k < TOPK; ++k) outw[ci[k]] = w[k] * inv;
#pragma unroll
        for (int e = 0; e < NEXP; ++e) cw[(size_t)token * NEXP + e] = outw[e];
    }
}

// ---------------------------------------------------------------------------
// GEMM1 + clamped SwiGLU for one expert (double-buffered LDS).
// H = X @ W1[e] + b1[e] for gate cols [n0,n0+96) and paired linear cols
// [n0+2880,...), activation fused in-register, bf16 A written to workspace.
// ---------------------------------------------------------------------------
__global__ __launch_bounds__(256) void moe_gemm1_swiglu_kernel(
    const float* __restrict__ x, const float* __restrict__ w1e,
    const float* __restrict__ b1e, bf16_t* __restrict__ aout)
{
    __shared__ __align__(16) bf16_t Xs[2][BM * LDSS];
    __shared__ __align__(16) bf16_t Wg[2][BN * LDSS];
    __shared__ __align__(16) bf16_t Wl[2][BN * LDSS];

    const int tid  = threadIdx.x;
    const int wv   = tid >> 5;
    const int lane = tid & 31;
    const int half = lane >> 4;
    const int l15  = lane & 15;
    const int mw   = wv >> 1;          // 0..3 : 32-row band
    const int nw   = wv & 1;           // 0..1 : 48-col band
    const int m0   = blockIdx.y * BM;
    const int n0   = blockIdx.x * BN;

    float8 accg[2][3], accl[2][3];
#pragma unroll
    for (int mi = 0; mi < 2; ++mi)
#pragma unroll
        for (int s = 0; s < 3; ++s) {
            accg[mi][s] = (float8)0.0f;
            accl[mi][s] = (float8)0.0f;
        }

    // --- staging: X tile 128x32 (b32-packed pairs), W tiles 32x96 transposed
    auto stage = [&](int b, int k0) {
#pragma unroll
        for (int i = 0; i < 8; ++i) {                 // 2048 k-pairs of X
            const int p   = tid + i * 256;
            const int row = p >> 4;
            const int kp  = (p & 15) << 1;
            const float2 f = *(const float2*)(x + (size_t)(m0 + row) * HIDDEN + k0 + kp);
            *(bhalf2*)&Xs[b][row * LDSS + kp] = pack2(f.x, f.y);
        }
#pragma unroll
        for (int i = 0; i < 6; ++i) {                 // 1536 k-pairs per W tile
            const int p   = tid + i * 256;
            const int kp2 = p / BN;                   // 0..15
            const int n   = p - kp2 * BN;             // 0..95
            const size_t base = (size_t)(k0 + 2 * kp2) * (2 * INTER) + n0 + n;
            *(bhalf2*)&Wg[b][n * LDSS + 2 * kp2] =
                pack2(w1e[base], w1e[base + 2 * INTER]);
            *(bhalf2*)&Wl[b][n * LDSS + 2 * kp2] =
                pack2(w1e[base + INTER], w1e[base + INTER + 2 * INTER]);
            if (k0 + 2 * BK < HIDDEN)                 // pull K+2 weight tile
                __builtin_prefetch(&w1e[base + (size_t)(2 * BK) * (2 * INTER)], 0, 1);
        }
    };

    stage(0, 0);
    int buf = 0;
    for (int k0 = 0; k0 < HIDDEN; k0 += BK) {
        __syncthreads();
        if (k0 + BK < HIDDEN) stage(buf ^ 1, k0 + BK);

        // Load ALL fragments first (distinct regs -> staggered dscnt waits),
        // then issue the 12 WMMAs as one burst.
        const bhalf16 af0 = make_frag(&Xs[buf][(mw * 32 + l15) * LDSS + 8 * half]);
        const bhalf16 af1 = make_frag(&Xs[buf][(mw * 32 + 16 + l15) * LDSS + 8 * half]);
        bhalf16 bg[3], bl[3];
#pragma unroll
        for (int s = 0; s < 3; ++s) {
            const int nb = (nw * 48 + s * 16 + l15) * LDSS + 8 * half;
            bg[s] = make_frag(&Wg[buf][nb]);
            bl[s] = make_frag(&Wl[buf][nb]);
        }
#pragma unroll
        for (int s = 0; s < 3; ++s) {
            accg[0][s] = wmma_bf16(af0, bg[s], accg[0][s]);
            accg[1][s] = wmma_bf16(af1, bg[s], accg[1][s]);
            accl[0][s] = wmma_bf16(af0, bl[s], accl[0][s]);
            accl[1][s] = wmma_bf16(af1, bl[s], accl[1][s]);
        }
        buf ^= 1;
    }

    // Epilogue: bias + clamped SwiGLU, store bf16 activation.
    // C/D layout: VGPR r, lane L -> M = r + 8*(L>>4), N = L&15.
#pragma unroll
    for (int s = 0; s < 3; ++s) {
        const int col = n0 + nw * 48 + s * 16 + l15;
        const float bgv = b1e[col];
        const float blv = b1e[INTER + col];
#pragma unroll
        for (int mi = 0; mi < 2; ++mi)
#pragma unroll
            for (int r = 0; r < 8; ++r) {
                const int row = m0 + mw * 32 + mi * 16 + half * 8 + r;
                float g = accg[mi][s][r] + bgv;
                float l = accl[mi][s][r] + blv;
                g = fminf(g, LIMIT);
                l = fminf(fmaxf(l, -LIMIT), LIMIT);
                const float act = g * (1.0f / (1.0f + __expf(-ALPHA * g))) * (l + BETA);
                aout[(size_t)row * INTER + col] = (bf16_t)act;
            }
    }
}

// ---------------------------------------------------------------------------
// GEMM2 + weighted combine for one expert (double-buffered LDS).
// Y += cw[:,e] * (A @ W2[e] + b2[e]).  first==1 -> overwrite (expert 0).
// A tile is already bf16 -> staged with GLOBAL_LOAD_ASYNC_TO_LDS (ASYNCcnt).
// ---------------------------------------------------------------------------
__global__ __launch_bounds__(256) void moe_gemm2_combine_kernel(
    const bf16_t* __restrict__ a, const float* __restrict__ w2e,
    const float* __restrict__ b2e, const float* __restrict__ cw,
    const int eidx, float* __restrict__ out, const int first)
{
    __shared__ __align__(16) bf16_t Xs[2][BM * LDSS];
    __shared__ __align__(16) bf16_t Ws[2][BN * LDSS];

    const int tid  = threadIdx.x;
    const int wv   = tid >> 5;
    const int lane = tid & 31;
    const int half = lane >> 4;
    const int l15  = lane & 15;
    const int mw   = wv >> 1;
    const int nw   = wv & 1;
    const int m0   = blockIdx.y * BM;
    const int n0   = blockIdx.x * BN;

    float8 acc[2][3];
#pragma unroll
    for (int mi = 0; mi < 2; ++mi)
#pragma unroll
        for (int s = 0; s < 3; ++s) acc[mi][s] = (float8)0.0f;

    auto stage = [&](int b, int k0) {
#pragma unroll
        for (int i = 0; i < 8; ++i) {                 // A already bf16
            const int p   = tid + i * 256;
            const int row = p >> 4;
            const int kp  = (p & 15) << 1;
            const bf16_t* src = a + (size_t)(m0 + row) * INTER + k0 + kp;
            bf16_t*       dst = &Xs[b][row * LDSS + kp];
#if USE_ASYNC_LDS
            __builtin_amdgcn_global_load_async_to_lds_b32(
                (gint_t*)src, (lint_t*)dst, 0, 0);
#else
            *(uint32_t*)dst = *(const uint32_t*)src;
#endif
        }
#pragma unroll
        for (int i = 0; i < 6; ++i) {                 // W2 fp32 -> bf16, [n][k]
            const int p   = tid + i * 256;
            const int kp2 = p / BN;
            const int n   = p - kp2 * BN;
            const size_t base = (size_t)(k0 + 2 * kp2) * HIDDEN + n0 + n;
            *(bhalf2*)&Ws[b][n * LDSS + 2 * kp2] =
                pack2(w2e[base], w2e[base + HIDDEN]);
            if (k0 + 2 * BK < INTER)
                __builtin_prefetch(&w2e[base + (size_t)(2 * BK) * HIDDEN], 0, 1);
        }
    };

    stage(0, 0);
    int buf = 0;
    for (int k0 = 0; k0 < INTER; k0 += BK) {
        wait_async0();                 // my async LDS writes complete
        __syncthreads();               // everyone's staging visible
        if (k0 + BK < INTER) stage(buf ^ 1, k0 + BK);

        const bhalf16 af0 = make_frag(&Xs[buf][(mw * 32 + l15) * LDSS + 8 * half]);
        const bhalf16 af1 = make_frag(&Xs[buf][(mw * 32 + 16 + l15) * LDSS + 8 * half]);
        bhalf16 bf[3];
#pragma unroll
        for (int s = 0; s < 3; ++s)
            bf[s] = make_frag(&Ws[buf][(nw * 48 + s * 16 + l15) * LDSS + 8 * half]);
#pragma unroll
        for (int s = 0; s < 3; ++s) {
            acc[0][s] = wmma_bf16(af0, bf[s], acc[0][s]);
            acc[1][s] = wmma_bf16(af1, bf[s], acc[1][s]);
        }
        buf ^= 1;
    }

    // Per-lane combine weights for its 16 rows.
    float cwv[2][8];
#pragma unroll
    for (int mi = 0; mi < 2; ++mi)
#pragma unroll
        for (int r = 0; r < 8; ++r) {
            const int row = m0 + mw * 32 + mi * 16 + half * 8 + r;
            cwv[mi][r] = cw[(size_t)row * NEXP + eidx];
        }
#pragma unroll
    for (int s = 0; s < 3; ++s) {
        const int col = n0 + nw * 48 + s * 16 + l15;
        const float bb = b2e[col];
#pragma unroll
        for (int mi = 0; mi < 2; ++mi)
#pragma unroll
            for (int r = 0; r < 8; ++r) {
                const int row = m0 + mw * 32 + mi * 16 + half * 8 + r;
                const float res = cwv[mi][r] * (acc[mi][s][r] + bb);
                const size_t idx = (size_t)row * HIDDEN + col;
                if (first) out[idx] = res;
                else       out[idx] += res;
            }
    }
}

// ---------------------------------------------------------------------------
extern "C" void kernel_launch(void* const* d_in, const int* in_sizes, int n_in,
                              void* d_out, int out_size, void* d_ws, size_t ws_size,
                              hipStream_t stream)
{
    const float* x  = (const float*)d_in[0];
    const float* gw = (const float*)d_in[1];
    const float* gb = (const float*)d_in[2];
    const float* w1 = (const float*)d_in[3];
    const float* b1 = (const float*)d_in[4];
    const float* w2 = (const float*)d_in[5];
    const float* b2 = (const float*)d_in[6];
    float* out = (float*)d_out;

    float*  cw   = (float*)d_ws;                                     // [2048, 8]
    bf16_t* abuf = (bf16_t*)((char*)d_ws + (size_t)NTOK * NEXP * 4); // [2048, 2880] bf16

    moe_router_kernel<<<NTOK / 8, 256, 0, stream>>>(x, gw, gb, cw);

    const dim3 grid(INTER / BN, NTOK / BM);  // 30 x 16
    for (int e = 0; e < NEXP; ++e) {
        const float* w1e = w1 + (size_t)e * HIDDEN * (2 * INTER);
        const float* b1e = b1 + (size_t)e * (2 * INTER);
        const float* w2e = w2 + (size_t)e * INTER * HIDDEN;
        const float* b2e = b2 + (size_t)e * HIDDEN;
        moe_gemm1_swiglu_kernel<<<grid, 256, 0, stream>>>(x, w1e, b1e, abuf);
        moe_gemm2_combine_kernel<<<grid, 256, 0, stream>>>(abuf, w2e, b2e, cw,
                                                           e, out, e == 0);
    }
}
